// Model_59837484368215
// MI455X (gfx1250) — matrix operand order repaired and checked by reading, not compiled, not run on hardware
//
#include <hip/hip_runtime.h>
#include <hip/hip_bf16.h>

// ---------------------------------------------------------------------------
// CDNA5 (gfx1250) hetero-GraphSAGE forward.
// GEMMs run on v_wmma_f32_16x16x32_bf16 (f32 accumulate), scatter/gather on
// float4 global ops + global_atomic_add_f32, everything launched on `stream`.
// ---------------------------------------------------------------------------

typedef __attribute__((ext_vector_type(16))) __bf16 v16bf;
typedef __attribute__((ext_vector_type(16))) float  v16f;
typedef __attribute__((ext_vector_type(8)))  float  v8f;

#define WPITCH 80  // bf16 elements per LDS B-row (160 B: 32B-aligned v16bf tiles)

// Native packed f32->bf16 (RNE): let clang lower a vector fptrunc.
__device__ __forceinline__ v16bf cvt16(v8f lo, v8f hi, float sc) {
  v16f vf;
#pragma unroll
  for (int i = 0; i < 8; i++) { vf[i] = lo[i] * sc; vf[8 + i] = hi[i] * sc; }
  return __builtin_convertvector(vf, v16bf);
}

#define WMMA_BF16(C, A, B) \
  __builtin_amdgcn_wmma_f32_16x16x32_bf16(false, (A), false, (B), (short)0, (C), false, false)

// ---------------------------------------------------------------------------
// Encoder: out[NT,64] = x_thesis[NT,384] @ lin_W.T + lin_b + emb_t[ids]
// ---------------------------------------------------------------------------
__global__ __launch_bounds__(256) void encode_thesis_kernel(
    const float* __restrict__ xth, const float* __restrict__ linW,
    const float* __restrict__ linb, const float* __restrict__ embt,
    const int* __restrict__ tids, float* __restrict__ out, int NT) {
  __shared__ alignas(32) __bf16 Wt[384 * WPITCH];   // B[k][n] = lin_W[n][k]
  const int t = threadIdx.x;
  for (int idx = t; idx < 64 * 384; idx += 256) {
    int n = idx / 384, k = idx - n * 384;
    Wt[k * WPITCH + n] = (__bf16)linW[idx];
  }
  __syncthreads();

  const int lane = t & 31, half = lane >> 4, l16 = lane & 15;
  const int row0 = (blockIdx.x * 8 + (t >> 5)) * 16;
  if (row0 >= NT) return;

  // x_thesis is streamed exactly once (614 MB >> 192 MB L2): non-temporal
  // loads keep the L2-resident node tables from being evicted.
  const float* ap = xth + (long)(row0 + l16) * 384 + 8 * half;
  v8f c0 = {}, c1 = {}, c2 = {}, c3 = {};
#pragma unroll
  for (int k0 = 0; k0 < 384; k0 += 32) {
    v8f lo = __builtin_nontemporal_load((const v8f*)(ap + k0));
    v8f hi = __builtin_nontemporal_load((const v8f*)(ap + k0 + 16));
    v16bf a = cvt16(lo, hi, 1.0f);
    const __bf16* br = &Wt[(k0 + 8 * half) * WPITCH + l16];
    v16bf b0, b1, b2, b3;
#pragma unroll
    for (int s = 0; s < 16; s++) {
      int kk = (s < 8 ? s : s + 8) * WPITCH;
      b0[s] = br[kk]; b1[s] = br[kk + 16]; b2[s] = br[kk + 32]; b3[s] = br[kk + 48];
    }
    c0 = WMMA_BF16(c0, a, b0);
    c1 = WMMA_BF16(c1, a, b1);
    c2 = WMMA_BF16(c2, a, b2);
    c3 = WMMA_BF16(c3, a, b3);
  }
  const float b0s = linb[l16],      b1s = linb[16 + l16];
  const float b2s = linb[32 + l16], b3s = linb[48 + l16];
#pragma unroll
  for (int j = 0; j < 8; j++) {
    int m = row0 + j + 8 * half;
    int id = tids[m];
    const float* ep = embt + (long)id * 64 + l16;
    float* op = out + (long)m * 64 + l16;
    op[0]  = c0[j] + b0s + __builtin_nontemporal_load(ep + 0);
    op[16] = c1[j] + b1s + __builtin_nontemporal_load(ep + 16);
    op[32] = c2[j] + b2s + __builtin_nontemporal_load(ep + 32);
    op[48] = c3[j] + b3s + __builtin_nontemporal_load(ep + 48);
  }
}

// ---------------------------------------------------------------------------
// SAGE linear: out[N,64] = (agg/max(cnt,1)) @ Wl.T + x @ Wr.T + b  [+ReLU]
// Implemented as one K=128 GEMM: A = [agg*rs | x], B = [Wl ; Wr] in LDS.
// ---------------------------------------------------------------------------
__global__ __launch_bounds__(256) void sage_gemm_kernel(
    const float* __restrict__ agg, const float* __restrict__ cnt,
    const float* __restrict__ x, const float* __restrict__ Wl,
    const float* __restrict__ Wr, const float* __restrict__ bias,
    float* __restrict__ out, int N, int relu) {
  __shared__ alignas(32) __bf16 Wt[128 * WPITCH];
  const int t = threadIdx.x;
  for (int idx = t; idx < 64 * 128; idx += 256) {
    int n = idx >> 7, k = idx & 127;
    float w = (k < 64) ? Wl[n * 64 + k] : Wr[n * 64 + (k - 64)];
    Wt[k * WPITCH + n] = (__bf16)w;
  }
  __syncthreads();

  const int lane = t & 31, half = lane >> 4, l16 = lane & 15;
  const int row0 = (blockIdx.x * 8 + (t >> 5)) * 16;
  if (row0 >= N) return;

  const int arow = row0 + l16;
  const float rs = 1.0f / fmaxf(cnt[arow], 1.0f);   // segment-mean folded in
  const float* gp = agg + (long)arow * 64 + 8 * half;
  const float* xp = x   + (long)arow * 64 + 8 * half;

  v8f c0 = {}, c1 = {}, c2 = {}, c3 = {};
#pragma unroll
  for (int k0 = 0; k0 < 128; k0 += 32) {
    const float* sp = (k0 < 64) ? (gp + k0) : (xp + (k0 - 64));
    const float  sc = (k0 < 64) ? rs : 1.0f;
    v8f lo = *(const v8f*)(sp);
    v8f hi = *(const v8f*)(sp + 16);
    v16bf a = cvt16(lo, hi, sc);
    const __bf16* br = &Wt[(k0 + 8 * half) * WPITCH + l16];
    v16bf b0, b1, b2, b3;
#pragma unroll
    for (int s = 0; s < 16; s++) {
      int kk = (s < 8 ? s : s + 8) * WPITCH;
      b0[s] = br[kk]; b1[s] = br[kk + 16]; b2[s] = br[kk + 32]; b3[s] = br[kk + 48];
    }
    c0 = WMMA_BF16(c0, a, b0);
    c1 = WMMA_BF16(c1, a, b1);
    c2 = WMMA_BF16(c2, a, b2);
    c3 = WMMA_BF16(c3, a, b3);
  }
  const float b0s = bias[l16],      b1s = bias[16 + l16];
  const float b2s = bias[32 + l16], b3s = bias[48 + l16];
#pragma unroll
  for (int j = 0; j < 8; j++) {
    int m = row0 + j + 8 * half;
    float v0 = c0[j] + b0s, v1 = c1[j] + b1s;
    float v2 = c2[j] + b2s, v3 = c3[j] + b3s;
    if (relu) {
      v0 = fmaxf(v0, 0.0f); v1 = fmaxf(v1, 0.0f);
      v2 = fmaxf(v2, 0.0f); v3 = fmaxf(v3, 0.0f);
    }
    float* op = out + (long)m * 64 + l16;
    op[0] = v0; op[16] = v1; op[32] = v2; op[48] = v3;
  }
}

// ---------------------------------------------------------------------------
// Helpers: zero fill, row gather, degree counts, edge scatter-add, edge dot
// ---------------------------------------------------------------------------
__global__ void zero_kernel(float4* p, long n4) {
  long stride = (long)gridDim.x * blockDim.x;
  for (long i = (long)blockIdx.x * blockDim.x + threadIdx.x; i < n4; i += stride)
    p[i] = make_float4(0.f, 0.f, 0.f, 0.f);
}

__global__ void gather_rows64(const float* __restrict__ src,
                              const int* __restrict__ ids,
                              float* __restrict__ out, int N) {
  long i = (long)blockIdx.x * blockDim.x + threadIdx.x;   // N*16 float4 slots
  if (i >= (long)N * 16) return;
  int row = (int)(i >> 4), c = (int)(i & 15);
  ((float4*)(out + (long)row * 64))[c] =
      ((const float4*)(src + (long)ids[row] * 64))[c];
}

__global__ void count_kernel(const int* __restrict__ src, const int* __restrict__ dst,
                             float* __restrict__ cnt_t, float* __restrict__ cnt_m, int E) {
  int e = blockIdx.x * blockDim.x + threadIdx.x;
  if (e >= E) return;
  atomicAdd(&cnt_m[dst[e]], 1.0f);
  atomicAdd(&cnt_t[src[e]], 1.0f);
}

// Both directions in one edge pass: gm[dst] += xt[src], gt[src] += xm[dst].
__global__ void agg_kernel(const float* __restrict__ xt, const float* __restrict__ xm,
                           const int* __restrict__ src, const int* __restrict__ dst,
                           float* __restrict__ gm, float* __restrict__ gt, int E) {
  long tid = (long)blockIdx.x * blockDim.x + threadIdx.x;
  long e = tid >> 4;
  if (e >= E) return;
  int c = (int)(tid & 15);
  int s = src[e], d = dst[e];
  float4 vt = ((const float4*)(xt + (long)s * 64))[c];
  float* pm = gm + (long)d * 64 + c * 4;
  atomicAdd(pm + 0, vt.x); atomicAdd(pm + 1, vt.y);
  atomicAdd(pm + 2, vt.z); atomicAdd(pm + 3, vt.w);
  float4 vm = ((const float4*)(xm + (long)d * 64))[c];
  float* pt = gt + (long)s * 64 + c * 4;
  atomicAdd(pt + 0, vm.x); atomicAdd(pt + 1, vm.y);
  atomicAdd(pt + 2, vm.z); atomicAdd(pt + 3, vm.w);
}

// out[e] = dot64(xt[el_src[e]], xm[el_dst[e]]) ; 16 lanes per edge.
__global__ void dot_kernel(const float* __restrict__ xt, const float* __restrict__ xm,
                           const int* __restrict__ es, const int* __restrict__ ed,
                           float* __restrict__ out, int EL) {
  long tid = (long)blockIdx.x * blockDim.x + threadIdx.x;
  long e = tid >> 4;
  if (e >= EL) return;
  int c = (int)(tid & 15);
  int s = es[e], d = ed[e];
  float4 a = ((const float4*)(xt + (long)s * 64))[c];
  float4 b = ((const float4*)(xm + (long)d * 64))[c];
  float p = a.x * b.x + a.y * b.y + a.z * b.z + a.w * b.w;
  p += __shfl_xor(p, 1, 32);
  p += __shfl_xor(p, 2, 32);
  p += __shfl_xor(p, 4, 32);
  p += __shfl_xor(p, 8, 32);
  if (c == 0) out[e] = p;
}

// ---------------------------------------------------------------------------
extern "C" void kernel_launch(void* const* d_in, const int* in_sizes, int n_in,
                              void* d_out, int out_size, void* d_ws, size_t ws_size,
                              hipStream_t stream) {
  const float* x_thesis = (const float*)d_in[0];
  const int*   tid_map  = (const int*)d_in[1];
  const int*   mid_map  = (const int*)d_in[2];
  const int*   edge_src = (const int*)d_in[3];
  const int*   edge_dst = (const int*)d_in[4];
  const int*   el_src   = (const int*)d_in[5];
  const int*   el_dst   = (const int*)d_in[6];
  const float* lin_W    = (const float*)d_in[7];
  const float* lin_b    = (const float*)d_in[8];
  const float* emb_t    = (const float*)d_in[9];
  const float* emb_m    = (const float*)d_in[10];
  const float* Wl_tm0 = (const float*)d_in[11], *Wr_tm0 = (const float*)d_in[12];
  const float* b_tm0  = (const float*)d_in[13];
  const float* Wl_mt0 = (const float*)d_in[14], *Wr_mt0 = (const float*)d_in[15];
  const float* b_mt0  = (const float*)d_in[16];
  const float* Wl_tm1 = (const float*)d_in[17], *Wr_tm1 = (const float*)d_in[18];
  const float* b_tm1  = (const float*)d_in[19];
  const float* Wl_mt1 = (const float*)d_in[20], *Wr_mt1 = (const float*)d_in[21];
  const float* b_mt1  = (const float*)d_in[22];

  const int NT = in_sizes[1], NM = in_sizes[2], E = in_sizes[3], EL = in_sizes[5];

  // Workspace layout (floats): xt_a|xt_b|xm_a|xm_b|gt|gm|cnt_t|cnt_m  (~340 MB)
  float* ws    = (float*)d_ws;
  float* xt_a  = ws;
  float* xt_b  = xt_a + (size_t)NT * 64;
  float* xm_a  = xt_b + (size_t)NT * 64;
  float* xm_b  = xm_a + (size_t)NM * 64;
  float* gt    = xm_b + (size_t)NM * 64;     // agg sums, thesis side
  float* gm    = gt   + (size_t)NT * 64;     // agg sums, mentor side
  float* cnt_t = gm   + (size_t)NM * 64;
  float* cnt_m = cnt_t + (size_t)NT;

  const dim3 B(256);
  const long gN = (long)NT * 64 + (long)NM * 64;  // contiguous gt..gm region

  // node encoders
  encode_thesis_kernel<<<(NT + 127) / 128, B, 0, stream>>>(
      x_thesis, lin_W, lin_b, emb_t, tid_map, xt_a, NT);
  gather_rows64<<<(int)(((long)NM * 16 + 255) / 256), B, 0, stream>>>(
      emb_m, mid_map, xm_a, NM);

  // degree counts (shared by both layers)
  zero_kernel<<<512, B, 0, stream>>>((float4*)cnt_t, (long)(NT + NM) / 4);
  count_kernel<<<(E + 255) / 256, B, 0, stream>>>(edge_src, edge_dst, cnt_t, cnt_m, E);

  // ---- layer 0 (ReLU) ----
  zero_kernel<<<2048, B, 0, stream>>>((float4*)gt, gN / 4);
  agg_kernel<<<(int)(((long)E * 16 + 255) / 256), B, 0, stream>>>(
      xt_a, xm_a, edge_src, edge_dst, gm, gt, E);
  sage_gemm_kernel<<<(NM + 127) / 128, B, 0, stream>>>(
      gm, cnt_m, xm_a, Wl_tm0, Wr_tm0, b_tm0, xm_b, NM, 1);
  sage_gemm_kernel<<<(NT + 127) / 128, B, 0, stream>>>(
      gt, cnt_t, xt_a, Wl_mt0, Wr_mt0, b_mt0, xt_b, NT, 1);

  // ---- layer 1 (no activation) ----
  zero_kernel<<<2048, B, 0, stream>>>((float4*)gt, gN / 4);
  agg_kernel<<<(int)(((long)E * 16 + 255) / 256), B, 0, stream>>>(
      xt_b, xm_b, edge_src, edge_dst, gm, gt, E);
  sage_gemm_kernel<<<(NM + 127) / 128, B, 0, stream>>>(
      gm, cnt_m, xm_b, Wl_tm1, Wr_tm1, b_tm1, xm_a, NM, 0);
  sage_gemm_kernel<<<(NT + 127) / 128, B, 0, stream>>>(
      gt, cnt_t, xt_b, Wl_mt1, Wr_mt1, b_mt1, xt_a, NT, 0);

  // edge dot-product classifier
  dot_kernel<<<(int)(((long)EL * 16 + 255) / 256), B, 0, stream>>>(
      xt_a, xm_a, el_src, el_dst, (float*)d_out, EL);
}
